// WindowAttention_35261681500738
// MI455X (gfx1250) — matrix operand order repaired and checked by reading, not compiled, not run on hardware
//
#include <hip/hip_runtime.h>
#include <hip/hip_bf16.h>

typedef _Float16 half_t;
typedef __attribute__((ext_vector_type(16))) _Float16 v16h;
typedef __attribute__((ext_vector_type(8)))  _Float16 v8h;
typedef __attribute__((ext_vector_type(4)))  _Float16 v4h;
typedef __attribute__((ext_vector_type(8)))  float    v8f;

#define B_WIN 512
#define NTOK  64
#define CDIM  512
#define NH    16
#define HD    32
#define MTOT  (B_WIN * NTOK)      // 32768 rows
#define QKVC  (3 * CDIM)          // 1536 qkv columns

// ---------------------------------------------------------------------------
// Fragment loaders. wave32; g = lane>>4 (half-wave), lid = lane&15.
//
// A operand (16x32 f16, M x K): lane (g,lid) = row M=lid,
//   elems 0..15 -> K = g*16 + i  (same K order as B).
// B operand (32x16 f16, K x N), "row striped across lanes": lane (g,lid) =
//   col N=lid, elems 0..15 -> K = g*16 + i.  (column must be K-contiguous)
// C/D (16x16 f32): lane (g,lid): vgpr j -> (M = j + 8*g, N = lid).
// ---------------------------------------------------------------------------
static __device__ __forceinline__ v16h make_frag(v8h lo, v8h hi) {
  return __builtin_shufflevector(lo, hi, 0,1,2,3,4,5,6,7,8,9,10,11,12,13,14,15);
}
static __device__ __forceinline__ v16h ldA(const half_t* row, int k0, int g) {
  const half_t* p = row + k0 + g * 16;                // 16B aligned
  return make_frag(*reinterpret_cast<const v8h*>(p),
                   *reinterpret_cast<const v8h*>(p + 8));
}
static __device__ __forceinline__ v16h ldB(const half_t* row, int k0, int g) {
  const half_t* p = row + k0 + g * 16;                // 16B aligned
  return make_frag(*reinterpret_cast<const v8h*>(p),
                   *reinterpret_cast<const v8h*>(p + 8));
}
static __device__ __forceinline__ v8f wmma_f16(v16h a, v16h b, v8f c) {
  return __builtin_amdgcn_wmma_f32_16x16x32_f16(false, a, false, b,
                                                (short)0, c, false, false);
}

// ---------------------------------------------------------------------------
// f32 -> f16 convert, 4 elements/thread
// ---------------------------------------------------------------------------
__global__ void __launch_bounds__(256)
cvt_f32_to_f16_x4(const float* __restrict__ in, half_t* __restrict__ out, int n4) {
  int i = blockIdx.x * blockDim.x + threadIdx.x;
  if (i < n4) {
    const float4 f = reinterpret_cast<const float4*>(in)[i];
    v4h h;
    h[0] = (half_t)f.x; h[1] = (half_t)f.y; h[2] = (half_t)f.z; h[3] = (half_t)f.w;
    reinterpret_cast<v4h*>(out)[i] = h;
  }
}

// ---------------------------------------------------------------------------
// QKV GEMM: [32768 x 512] x [512 x 1536] + bias.
// Each wave: 32(M) x 64(N) tile -> 8 WMMA per K-step, B frags reused 2x.
// Block 128 thr = 4 waves = 128x64 block tile. grid (256, 24).
// Epilogue scatters into attention-friendly layouts:
//   q: qh[b][h][n][d] (pre-scaled by hd^-0.5), k: kh[b][h][n][d],
//   v: vT[b][h][d][n]  (transposed so P@V needs K-contiguous columns).
// ---------------------------------------------------------------------------
__global__ void __launch_bounds__(128)
qkv_gemm_kernel(const half_t* __restrict__ xh,   // [MTOT][CDIM]
                const half_t* __restrict__ wh,   // [QKVC][CDIM] (qkv_w rows)
                const float*  __restrict__ qkvb, // [QKVC]
                half_t* __restrict__ qh, half_t* __restrict__ kh,
                half_t* __restrict__ vT) {
  const int wave = threadIdx.x >> 5;
  const int lane = threadIdx.x & 31;
  const int g    = lane >> 4;
  const int lid  = lane & 15;
  const int mbase = blockIdx.x * 128 + wave * 32;
  const int nbase = blockIdx.y * 64;

  const half_t* arow0 = xh + (size_t)(mbase + lid) * CDIM;
  const half_t* arow1 = arow0 + (size_t)16 * CDIM;
  const half_t* brow0 = wh + (size_t)(nbase + lid) * CDIM;

  const v8f z = {0.f,0.f,0.f,0.f,0.f,0.f,0.f,0.f};
  v8f acc0[4] = {z, z, z, z};
  v8f acc1[4] = {z, z, z, z};

  for (int k0 = 0; k0 < CDIM; k0 += 32) {
    __builtin_prefetch(arow0 + k0 + 64, 0, 3);   // near-scope global_prefetch_b8
    __builtin_prefetch(arow1 + k0 + 64, 0, 3);
    v16h a0 = ldA(arow0, k0, g);
    v16h a1 = ldA(arow1, k0, g);
#pragma unroll
    for (int t = 0; t < 4; ++t) {
      v16h b = ldB(brow0 + (size_t)t * 16 * CDIM, k0, g);
      acc0[t] = wmma_f16(a0, b, acc0[t]);
      acc1[t] = wmma_f16(a1, b, acc1[t]);
    }
  }

  const float scale = 0.17677669529663688f;     // 1/sqrt(32)
#pragma unroll
  for (int t = 0; t < 4; ++t) {
    const int r  = nbase + t * 16 + lid;        // output column in [0,1536)
    const float bv = qkvb[r];
    const int which = r >> 9;                   // 0=q 1=k 2=v (uniform per tile)
    const int h = (r >> 5) & (NH - 1);
    const int d = r & (HD - 1);
#pragma unroll
    for (int ms = 0; ms < 2; ++ms) {
#pragma unroll
      for (int j = 0; j < 8; ++j) {
        const int mrow = mbase + ms * 16 + g * 8 + j;   // global row = b*64 + n
        const int b = mrow >> 6;
        const int n = mrow & 63;
        const float val = (ms ? acc1[t][j] : acc0[t][j]) + bv;
        if (which == 0)
          qh[(((size_t)b * NH + h) * NTOK + n) * HD + d] = (half_t)(val * scale);
        else if (which == 1)
          kh[(((size_t)b * NH + h) * NTOK + n) * HD + d] = (half_t)val;
        else
          vT[(((size_t)b * NH + h) * HD + d) * NTOK + n] = (half_t)val;
      }
    }
  }
}

// ---------------------------------------------------------------------------
// Attention: one block (128 thr, 4 waves) per (b,h).
//   S = q k^T (+bias+mask)  -> fp32 LDS -> softmax -> f16 P -> O = P vT
//   Output oh[b][n][h*32+d] (row-major [MTOT][CDIM] for proj GEMM).
// ---------------------------------------------------------------------------
#define SSTRIDE 65   // fp32, breaks 64-bank alias
#define PSTRIDE 72   // f16, 144B rows (16B aligned), spreads banks

__global__ void __launch_bounds__(128)
attn_kernel(const half_t* __restrict__ qh, const half_t* __restrict__ kh,
            const half_t* __restrict__ vT, const float* __restrict__ attn_bias,
            const float* __restrict__ mask, half_t* __restrict__ oh) {
  __shared__ float  S[NTOK * SSTRIDE];
  __shared__ half_t P[NTOK * PSTRIDE];

  const int bh = blockIdx.x;
  const int b  = bh >> 4;
  const int h  = bh & (NH - 1);
  const int wave = threadIdx.x >> 5;
  const int lane = threadIdx.x & 31;
  const int g    = lane >> 4;
  const int lid  = lane & 15;
  const int mrow = wave * 16;

  const half_t* qb = qh + (size_t)bh * NTOK * HD;
  const half_t* kb = kh + (size_t)bh * NTOK * HD;
  const half_t* vb = vT + (size_t)bh * HD * NTOK;
  const float* biasb = attn_bias + (size_t)h * NTOK * NTOK;
  const float* maskb = mask      + (size_t)b * NTOK * NTOK;

  // ---- S = (q*scale) @ k^T : K = hd = 32, single WMMA per 16x16 tile ----
  const v8f z = {0.f,0.f,0.f,0.f,0.f,0.f,0.f,0.f};
  {
    v16h a = ldA(qb + (size_t)(mrow + lid) * HD, 0, g);
#pragma unroll
    for (int t = 0; t < 4; ++t) {
      v16h bf = ldB(kb + (size_t)(t * 16 + lid) * HD, 0, g);
      v8f s = wmma_f16(a, bf, z);
#pragma unroll
      for (int j = 0; j < 8; ++j) {
        const int m = mrow + g * 8 + j;
        const int n = t * 16 + lid;
        S[m * SSTRIDE + n] = s[j] + biasb[m * 64 + n] + maskb[m * 64 + n];
      }
    }
  }
  __syncthreads();

  // ---- fp32 softmax per row, emit f16 P ----
  if (threadIdx.x < NTOK) {
    float* Srow = &S[threadIdx.x * SSTRIDE];
    float mx = Srow[0];
#pragma unroll
    for (int i = 1; i < 64; ++i) mx = fmaxf(mx, Srow[i]);
    float sum = 0.f;
#pragma unroll
    for (int i = 0; i < 64; ++i) { float e = __expf(Srow[i] - mx); Srow[i] = e; sum += e; }
    const float inv = 1.0f / sum;
    half_t* Prow = &P[threadIdx.x * PSTRIDE];
#pragma unroll
    for (int i = 0; i < 64; ++i) Prow[i] = (half_t)(Srow[i] * inv);
  }
  __syncthreads();

  // ---- O = P @ vT : K = 64 (two WMMA steps), N = hd = 32 (two tiles) ----
  v8f o[2] = {z, z};
#pragma unroll
  for (int k0 = 0; k0 < 64; k0 += 32) {
    v16h pa = ldA(&P[(mrow + lid) * PSTRIDE], k0, g);   // ds_load_b128 x2
#pragma unroll
    for (int t = 0; t < 2; ++t) {
      v16h vf = ldB(vb + (size_t)(t * 16 + lid) * NTOK, k0, g);
      o[t] = wmma_f16(pa, vf, o[t]);
    }
  }
#pragma unroll
  for (int t = 0; t < 2; ++t) {
#pragma unroll
    for (int j = 0; j < 8; ++j) {
      const int m = mrow + g * 8 + j;      // token
      const int d = t * 16 + lid;
      oh[((size_t)b * NTOK + m) * CDIM + h * HD + d] = (half_t)o[t][j];
    }
  }
}

// ---------------------------------------------------------------------------
// Proj GEMM: [32768 x 512] x [512 x 512] + bias -> fp32 out.
// Each wave: 32x64 tile. grid (256, 8), block 128.
// ---------------------------------------------------------------------------
__global__ void __launch_bounds__(128)
proj_gemm_kernel(const half_t* __restrict__ oh,  // [MTOT][CDIM]
                 const half_t* __restrict__ pwh, // [CDIM][CDIM] (proj_w rows)
                 const float*  __restrict__ pb,  // [CDIM]
                 float* __restrict__ out) {      // [MTOT][CDIM]
  const int wave = threadIdx.x >> 5;
  const int lane = threadIdx.x & 31;
  const int g    = lane >> 4;
  const int lid  = lane & 15;
  const int mbase = blockIdx.x * 128 + wave * 32;
  const int nbase = blockIdx.y * 64;

  const half_t* arow0 = oh  + (size_t)(mbase + lid) * CDIM;
  const half_t* arow1 = arow0 + (size_t)16 * CDIM;
  const half_t* brow0 = pwh + (size_t)(nbase + lid) * CDIM;

  const v8f z = {0.f,0.f,0.f,0.f,0.f,0.f,0.f,0.f};
  v8f acc0[4] = {z, z, z, z};
  v8f acc1[4] = {z, z, z, z};

  for (int k0 = 0; k0 < CDIM; k0 += 32) {
    __builtin_prefetch(arow0 + k0 + 64, 0, 3);
    __builtin_prefetch(arow1 + k0 + 64, 0, 3);
    v16h a0 = ldA(arow0, k0, g);
    v16h a1 = ldA(arow1, k0, g);
#pragma unroll
    for (int t = 0; t < 4; ++t) {
      v16h b = ldB(brow0 + (size_t)t * 16 * CDIM, k0, g);
      acc0[t] = wmma_f16(a0, b, acc0[t]);
      acc1[t] = wmma_f16(a1, b, acc1[t]);
    }
  }
#pragma unroll
  for (int t = 0; t < 4; ++t) {
    const int n = nbase + t * 16 + lid;
    const float bv = pb[n];
#pragma unroll
    for (int ms = 0; ms < 2; ++ms) {
#pragma unroll
      for (int j = 0; j < 8; ++j) {
        const int m = mbase + ms * 16 + g * 8 + j;
        out[(size_t)m * CDIM + n] = (ms ? acc1[t][j] : acc0[t][j]) + bv;
      }
    }
  }
}

// ---------------------------------------------------------------------------
extern "C" void kernel_launch(void* const* d_in, const int* in_sizes, int n_in,
                              void* d_out, int out_size, void* d_ws, size_t ws_size,
                              hipStream_t stream) {
  const float* x       = (const float*)d_in[0];
  const float* bias_in = (const float*)d_in[1];
  const float* mask    = (const float*)d_in[2];
  const float* qkv_w   = (const float*)d_in[3];
  const float* qkv_b   = (const float*)d_in[4];
  const float* proj_w  = (const float*)d_in[5];
  const float* proj_b  = (const float*)d_in[6];
  float* out = (float*)d_out;

  const size_t nX = (size_t)MTOT * CDIM;       // 16,777,216 elems
  half_t* xh  = (half_t*)d_ws;                 // f16 x; reused as oh after QKV
  half_t* qh  = xh  + nX;
  half_t* kh  = qh  + nX;
  half_t* vT  = kh  + nX;
  half_t* wqh = vT  + nX;                      // qkv_w f16: 786,432
  half_t* wph = wqh + (size_t)QKVC * CDIM;     // proj_w f16: 262,144
  half_t* oh  = xh;                            // alias: xh dead after QKV GEMM

  // 1) f32 -> f16 conversions
  cvt_f32_to_f16_x4<<<dim3((unsigned)(nX / 4 / 256)), 256, 0, stream>>>(x, xh, (int)(nX / 4));
  cvt_f32_to_f16_x4<<<dim3((QKVC * CDIM / 4) / 256), 256, 0, stream>>>(qkv_w, wqh, QKVC * CDIM / 4);
  cvt_f32_to_f16_x4<<<dim3((CDIM * CDIM / 4) / 256), 256, 0, stream>>>(proj_w, wph, CDIM * CDIM / 4);

  // 2) QKV projection (WMMA), scatter to q/k/vT attention layouts
  qkv_gemm_kernel<<<dim3(MTOT / 128, QKVC / 64), 128, 0, stream>>>(xh, wqh, qkv_b, qh, kh, vT);

  // 3) Attention per (b,h): S = qk^T + bias + mask, softmax, O = P vT
  attn_kernel<<<dim3(B_WIN * NH), 128, 0, stream>>>(qh, kh, vT, bias_in, mask, oh);

  // 4) Output projection (WMMA) -> fp32
  proj_gemm_kernel<<<dim3(MTOT / 128, CDIM / 64), 128, 0, stream>>>(oh, wph, proj_b, out);
}
